// MBGCN_51127290691695
// MI455X (gfx1250) — matrix operand order repaired and checked by reading, not compiled, not run on hardware
//
#include <hip/hip_runtime.h>
#include <hip/hip_bf16.h>

// ---------------------------------------------------------------------------
// MBGCN scoring on MI455X (gfx1250, wave32).
// SpMMs: fp32 atomic scatter (L2-resident operands, atomic-throughput bound).
// Dense transforms: v_wmma_f32_16x16x32_bf16 with fp32 accumulate; operands
// staged in LDS pre-swizzled into the ISA 7.12.2 fragment layout so every
// fragment load is two contiguous ds_load_b128 (no scalar u16 packing).
// Gathered-row optimization: proj/item_prop computed only for the 4096 rows
// actually consumed (25x less GEMM work than the reference graph).
// ---------------------------------------------------------------------------

#define EPSF 1e-8f

typedef __attribute__((ext_vector_type(16))) __bf16 v16bf;
typedef __attribute__((ext_vector_type(8)))  float  v8f;

__device__ inline v8f wmma_bf16(v16bf a, v16bf b, v8f c) {
  return __builtin_amdgcn_wmma_f32_16x16x32_bf16(
      /*neg_a=*/false, a, /*neg_b=*/false, b,
      /*c_mod=*/(short)0, c, /*reuse_a=*/false, /*reuse_b=*/false);
}

// Inverse maps: fragment-linear index -> (row, k) of the source matrix.
// A fragment (16xK, 16-bit): lane = m + 16*(k-half); per lane 16 contiguous
// bf16 in LDS. i layout: [ks][lane][j], j fastest.
__device__ inline void a_frag_coords(int i, int& m, int& k) {
  int ks = i >> 9;                 // 512 elements per 16x32 k-step
  int t  = i & 511;
  int lane = t >> 4, j = t & 15;
  m = lane & 15;
  int kh = (lane >> 4) << 3;       // 0 or 8
  int p  = j >> 1;
  int kb = (p < 4) ? (p * 2) : (16 + (p - 4) * 2);
  k = ks * 32 + kb + kh + (j & 1);
}

// B fragment (KxN weights, 16-bit): lanes 0-15 K=0..7,16..23, lanes 16-31 K=8..15,24..31.
// i layout: [ks][nt][lane][j], j fastest.  NUMNT must be a literal at call.
__device__ inline void b_frag_coords(int i, int numNT, int& n, int& k) {
  int perKS = numNT << 9;
  int ks = i / perKS;
  int t  = i - ks * perKS;
  int nt = t >> 9;
  int t2 = t & 511;
  int lane = t2 >> 4, j = t2 & 15;
  n = nt * 16 + (lane & 15);
  k = ks * 32 + ((j & 8) << 1) + ((lane >> 4) << 3) + (j & 7);
}

__device__ inline v16bf frag_ld(const __bf16* p) {
  return *(const v16bf*)p;         // 32B contiguous -> 2x ds_load_b128
}

// ---------------------------------------------------------------------------
__global__ void zero_f32(float* __restrict__ p, long n) {
  long i = (long)blockIdx.x * blockDim.x + threadIdx.x;
  if (i < n) p[i] = 0.0f;
}

// COO SpMM scatter: out[rows[nz], :] += vals[nz] * dense[cols[nz], :]
// One thread per (nnz, 4 columns): global_load_b128 + 4 f32 atomics.
__global__ void spmm_atomic(const int* __restrict__ rows,
                            const int* __restrict__ cols,
                            const float* __restrict__ vals,
                            const float* __restrict__ dense,
                            float* __restrict__ out,
                            int nnz, int qE, int E) {
  long idx = (long)blockIdx.x * blockDim.x + threadIdx.x;
  if (idx >= (long)nnz * qE) return;
  int nz = (int)(idx / qE);
  int e  = (int)(idx - (long)nz * qE) * 4;
  int rr = rows[nz];
  int cc = cols[nz];
  float v = vals[nz];
  const float4 d = *(const float4*)(dense + (long)cc * E + e);
  float* o = out + (long)rr * E + e;
  atomicAdd(o + 0, v * d.x);
  atomicAdd(o + 1, v * d.y);
  atomicAdd(o + 2, v * d.z);
  atomicAdd(o + 3, v * d.w);
}

// tip_cat[i, 0:64]   = item_embedding[i, :]
// tip_cat[i, 64:128] = ((neigh[i,:] / (deg[i]+eps)) @ Wp)   (K=64, N=64)
__global__ __launch_bounds__(128) void tip_cat_wmma(
    const float* __restrict__ neigh, const float* __restrict__ deg,
    const float* __restrict__ Wp, const float* __restrict__ item_emb,
    float* __restrict__ tip_cat, int I) {
  __shared__ __align__(32) __bf16 sA[2 * 512];     // 2 k-steps
  __shared__ __align__(32) __bf16 sB[2 * 4 * 512]; // 2 k-steps x 4 n-tiles
  __shared__ float sInv[16];
  const int row0 = blockIdx.x * 16;
  const int tid  = threadIdx.x;

  if (tid < 16) {
    int r = row0 + tid; r = (r < I) ? r : (I - 1);
    sInv[tid] = 1.0f / (deg[r] + EPSF);
  }
  __syncthreads();

  for (int i = tid; i < 2 * 4 * 512; i += 128) {
    int n, k; b_frag_coords(i, 4, n, k);
    sB[i] = (__bf16)Wp[k * 64 + n];
  }
  for (int i = tid; i < 2 * 512; i += 128) {
    int m, k; a_frag_coords(i, m, k);
    int r = row0 + m; r = (r < I) ? r : (I - 1);
    sA[i] = (__bf16)(neigh[(long)r * 64 + k] * sInv[m]);
  }
  __syncthreads();

  const int wave = tid >> 5, lane = tid & 31;      // 4 waves cover N=64
  v8f c = {};
#pragma unroll
  for (int ks = 0; ks < 2; ++ks) {
    v16bf a = frag_ld(sA + (ks * 32 + lane) * 16);
    v16bf b = frag_ld(sB + ((ks * 4 + wave) * 32 + lane) * 16);
    c = wmma_bf16(a, b, c);
  }
  const int n = lane & 15, mh = (lane >> 4) << 3;
  if (row0 + 16 <= I) {                            // uniform fast path
#pragma unroll
    for (int j = 0; j < 8; ++j)
      tip_cat[(long)(row0 + j + mh) * 128 + 64 + wave * 16 + n] = c[j];
  } else {
#pragma unroll
    for (int j = 0; j < 8; ++j) {
      int r = row0 + j + mh;
      if (r < I) tip_cat[(long)r * 128 + 64 + wave * 16 + n] = c[j];
    }
  }
  // copy embedding half (vectorized)
  for (int i = tid; i < 16 * 16; i += 128) {       // 16 rows x 16 float4
    int m = i >> 4, k4 = (i & 15) * 4;
    int r = row0 + m;
    if (r < I)
      *(float4*)(tip_cat + (long)r * 128 + k4) =
          *(const float4*)(item_emb + (long)r * 64 + k4);
  }
}

// Gathered behavior projection:
//   rowA = unp[user[b], :] / (ubd[user[b], rsel] + eps)      (K=128)
//   score2[b, :] += (rowA @ Wb) * tip_cat[item[b], :]        (N=128)
__global__ __launch_bounds__(256) void behavior_wmma(
    const float* __restrict__ unp, const float* __restrict__ ubd, int rsel,
    const float* __restrict__ Wb, const int* __restrict__ user,
    const int* __restrict__ item, const float* __restrict__ tip_cat,
    float* __restrict__ score2, int B) {
  __shared__ __align__(32) __bf16 sW[4 * 8 * 512];  // 32 KB
  __shared__ __align__(32) __bf16 sA[4 * 512];
  __shared__ int   sU[16];
  __shared__ float sInv[16];
  __shared__ int   sItem[16];
  const int row0 = blockIdx.x * 16;
  const int tid  = threadIdx.x;

  if (tid < 16) {
    int b = row0 + tid; b = (b < B) ? b : (B - 1);
    int u = user[b];
    sU[tid]    = u;
    sInv[tid]  = 1.0f / (ubd[(long)u * 3 + rsel] + EPSF);
    sItem[tid] = item[b];
  }
  __syncthreads();

  for (int i = tid; i < 4 * 8 * 512; i += 256) {
    int n, k; b_frag_coords(i, 8, n, k);
    sW[i] = (__bf16)Wb[k * 128 + n];
  }
  for (int i = tid; i < 4 * 512; i += 256) {
    int m, k; a_frag_coords(i, m, k);
    sA[i] = (__bf16)(unp[(long)sU[m] * 128 + k] * sInv[m]);
  }
  __syncthreads();

  const int wave = tid >> 5, lane = tid & 31;       // 8 waves cover N=128
  v8f c = {};
#pragma unroll
  for (int ks = 0; ks < 4; ++ks) {
    v16bf a = frag_ld(sA + (ks * 32 + lane) * 16);
    v16bf b = frag_ld(sW + ((ks * 8 + wave) * 32 + lane) * 16);
    c = wmma_bf16(a, b, c);
  }
  const int n = lane & 15, mh = (lane >> 4) << 3;
  if (row0 + 16 <= B) {
#pragma unroll
    for (int j = 0; j < 8; ++j) {
      int m = j + mh, col = wave * 16 + n;
      float t = tip_cat[(long)sItem[m] * 128 + col];
      score2[(long)(row0 + m) * 128 + col] += c[j] * t;  // r-loop serializes
    }
  } else {
#pragma unroll
    for (int j = 0; j < 8; ++j) {
      int m = j + mh, b = row0 + m;
      if (b < B) {
        int col = wave * 16 + n;
        float t = tip_cat[(long)sItem[m] * 128 + col];
        score2[(long)b * 128 + col] += c[j] * t;
      }
    }
  }
}

// out[b, 0:64] = item_ft[item[b], :] @ Wi   (gathered [B,64]@[64,64])
__global__ __launch_bounds__(128) void itemprop_wmma(
    const float* __restrict__ item_ft, const int* __restrict__ item,
    const float* __restrict__ Wi, float* __restrict__ out, int B) {
  __shared__ __align__(32) __bf16 sA[2 * 512];
  __shared__ __align__(32) __bf16 sB[2 * 4 * 512];
  __shared__ int sIt[16];
  const int row0 = blockIdx.x * 16;
  const int tid  = threadIdx.x;

  if (tid < 16) {
    int b = row0 + tid; b = (b < B) ? b : (B - 1);
    sIt[tid] = item[b];
  }
  __syncthreads();

  for (int i = tid; i < 2 * 4 * 512; i += 128) {
    int n, k; b_frag_coords(i, 4, n, k);
    sB[i] = (__bf16)Wi[k * 64 + n];
  }
  for (int i = tid; i < 2 * 512; i += 128) {
    int m, k; a_frag_coords(i, m, k);
    sA[i] = (__bf16)item_ft[(long)sIt[m] * 64 + k];
  }
  __syncthreads();

  const int wave = tid >> 5, lane = tid & 31;
  v8f c = {};
#pragma unroll
  for (int ks = 0; ks < 2; ++ks) {
    v16bf a = frag_ld(sA + (ks * 32 + lane) * 16);
    v16bf b = frag_ld(sB + ((ks * 4 + wave) * 32 + lane) * 16);
    c = wmma_bf16(a, b, c);
  }
  const int n = lane & 15, mh = (lane >> 4) << 3;
  if (row0 + 16 <= B) {
#pragma unroll
    for (int j = 0; j < 8; ++j)
      out[(long)(row0 + j + mh) * 64 + wave * 16 + n] = c[j];
  } else {
#pragma unroll
    for (int j = 0; j < 8; ++j) {
      int b = row0 + j + mh;
      if (b < B) out[(long)b * 64 + wave * 16 + n] = c[j];
    }
  }
}

// out[b, 0:64] = (score2[b,:]/3) @ Wu   ([B,128]@[128,64])
__global__ __launch_bounds__(128) void userprop_wmma(
    const float* __restrict__ score2, const float* __restrict__ Wu,
    float* __restrict__ out, int B) {
  __shared__ __align__(32) __bf16 sA[4 * 512];
  __shared__ __align__(32) __bf16 sB[4 * 4 * 512];  // 16 KB
  const int row0 = blockIdx.x * 16;
  const int tid  = threadIdx.x;
  const float invR = 1.0f / 3.0f;

  for (int i = tid; i < 4 * 4 * 512; i += 128) {
    int n, k; b_frag_coords(i, 4, n, k);
    sB[i] = (__bf16)Wu[k * 64 + n];
  }
  for (int i = tid; i < 4 * 512; i += 128) {
    int m, k; a_frag_coords(i, m, k);
    int b = row0 + m; b = (b < B) ? b : (B - 1);
    sA[i] = (__bf16)(score2[(long)b * 128 + k] * invR);
  }
  __syncthreads();

  const int wave = tid >> 5, lane = tid & 31;       // 4 waves cover N=64
  v8f c = {};
#pragma unroll
  for (int ks = 0; ks < 4; ++ks) {
    v16bf a = frag_ld(sA + (ks * 32 + lane) * 16);
    v16bf b = frag_ld(sB + ((ks * 4 + wave) * 32 + lane) * 16);
    c = wmma_bf16(a, b, c);
  }
  const int n = lane & 15, mh = (lane >> 4) << 3;
  if (row0 + 16 <= B) {
#pragma unroll
    for (int j = 0; j < 8; ++j)
      out[(long)(row0 + j + mh) * 64 + wave * 16 + n] = c[j];
  } else {
#pragma unroll
    for (int j = 0; j < 8; ++j) {
      int b = row0 + j + mh;
      if (b < B) out[(long)b * 64 + wave * 16 + n] = c[j];
    }
  }
}

// score1 dot, final scores (score1 + 0.5*score2/3) and L2 reduction.
__global__ __launch_bounds__(128) void final_score(
    const int* __restrict__ user, const int* __restrict__ item,
    const float* __restrict__ user_emb, const float* __restrict__ item_emb,
    const float* __restrict__ user_prop, const float* __restrict__ item_prop_g,
    const float* __restrict__ score2, float* __restrict__ out, int B) {
  const int b = blockIdx.x;
  const int c = threadIdx.x;          // 0..127
  const int u = user[b];
  const int it = item[b];

  float uf  = (c < 64) ? user_emb[(long)u * 64 + c]  : user_prop[(long)b * 64 + (c - 64)];
  float itf = (c < 64) ? item_emb[(long)it * 64 + c] : item_prop_g[(long)b * 64 + (c - 64)];

  __shared__ float red[4];
  float p = uf * itf;
#pragma unroll
  for (int off = 16; off > 0; off >>= 1) p += __shfl_down(p, off);
  if ((c & 31) == 0) red[c >> 5] = p;
  __syncthreads();
  float s1 = red[0] + red[1] + red[2] + red[3];

  out[(long)b * 128 + c] = s1 + 0.5f * (score2[(long)b * 128 + c] * (1.0f / 3.0f));

  float l2p = uf * uf + itf * itf;
#pragma unroll
  for (int off = 16; off > 0; off >>= 1) l2p += __shfl_down(l2p, off);
  if ((c & 31) == 0) atomicAdd(out + (long)B * 128, 1e-4f * l2p);
}

// ---------------------------------------------------------------------------
static inline void launch_zero(float* p, long n, hipStream_t s) {
  long blocks = (n + 255) / 256;
  zero_f32<<<(unsigned)blocks, 256, 0, s>>>(p, n);
}

static inline void launch_spmm(const int* rows, const int* cols, const float* vals,
                               const float* dense, float* out, int nnz, int E,
                               hipStream_t s) {
  int qE = E / 4;
  long total = (long)nnz * qE;
  long blocks = (total + 255) / 256;
  spmm_atomic<<<(unsigned)blocks, 256, 0, s>>>(rows, cols, vals, dense, out, nnz, qE, E);
}

extern "C" void kernel_launch(void* const* d_in, const int* in_sizes, int n_in,
                              void* d_out, int out_size, void* d_ws, size_t ws_size,
                              hipStream_t stream) {
  const int*   user     = (const int*)d_in[0];
  const int*   item     = (const int*)d_in[1];
  const int*   tm_rows  = (const int*)d_in[2];
  const int*   tm_cols  = (const int*)d_in[3];
  const float* tm_vals  = (const float*)d_in[4];
  const int*   rel_rows = (const int*)d_in[5];
  const int*   rel_cols = (const int*)d_in[6];
  const float* rel_vals = (const float*)d_in[7];
  const int*   ig_rows  = (const int*)d_in[8];
  const int*   ig_cols  = (const int*)d_in[9];
  const float* ig_vals  = (const float*)d_in[10];
  const float* ig_deg   = (const float*)d_in[11];
  const float* ubd      = (const float*)d_in[12];
  const float* user_emb = (const float*)d_in[13];
  const float* item_emb = (const float*)d_in[14];
  const float* Wp       = (const float*)d_in[15];   // [3,64,64]
  const float* Wb       = (const float*)d_in[16];   // [3,128,128]
  const float* Wu       = (const float*)d_in[17];   // [128,64]
  const float* Wi       = (const float*)d_in[18];   // [64,64]

  const int B       = in_sizes[0];
  const int U       = in_sizes[13] / 64;
  const int I       = in_sizes[14] / 64;
  const int NNZ_TM  = in_sizes[2];
  const int NNZ_REL = in_sizes[5] / 3;
  const int NNZ_IG  = in_sizes[8] / 3;
  const int R       = 3;

  // Workspace layout (fp32 elements)
  float* ws = (float*)d_ws;
  size_t off = 0;
  float* neigh     = ws + off; off += (size_t)I * 64;
  float* tip_cat   = ws + off; off += (size_t)I * 128;
  float* unp       = ws + off; off += (size_t)U * 128;
  float* item_ft   = ws + off; off += (size_t)I * 64;
  float* score2    = ws + off; off += (size_t)B * 128;
  float* user_prop = ws + off; off += (size_t)B * 64;
  float* item_prop = ws + off; off += (size_t)B * 64;
  (void)ws_size; (void)n_in; (void)out_size;

  float* out = (float*)d_out;

  // item_ft = tm^T @ user_embedding  (rows = tm_cols, cols = tm_rows)
  launch_zero(item_ft, (long)I * 64, stream);
  launch_spmm(tm_cols, tm_rows, tm_vals, user_emb, item_ft, NNZ_TM, 64, stream);

  launch_zero(score2, (long)B * 128, stream);

  for (int r = 0; r < R; ++r) {
    // item-item propagation
    launch_zero(neigh, (long)I * 64, stream);
    launch_spmm(ig_rows + (size_t)r * NNZ_IG, ig_cols + (size_t)r * NNZ_IG,
                ig_vals + (size_t)r * NNZ_IG, item_emb, neigh, NNZ_IG, 64, stream);
    tip_cat_wmma<<<(I + 15) / 16, 128, 0, stream>>>(
        neigh, ig_deg + (size_t)r * I, Wp + (size_t)r * 64 * 64, item_emb, tip_cat, I);

    // user aggregation over relation r
    launch_zero(unp, (long)U * 128, stream);
    launch_spmm(rel_rows + (size_t)r * NNZ_REL, rel_cols + (size_t)r * NNZ_REL,
                rel_vals + (size_t)r * NNZ_REL, tip_cat, unp, NNZ_REL, 128, stream);

    // gathered projection + fused product accumulate into score2
    behavior_wmma<<<(B + 15) / 16, 256, 0, stream>>>(
        unp, ubd, r, Wb + (size_t)r * 128 * 128, user, item, tip_cat, score2, B);
  }

  itemprop_wmma<<<(B + 15) / 16, 128, 0, stream>>>(item_ft, item, Wi, item_prop, B);
  userprop_wmma<<<(B + 15) / 16, 128, 0, stream>>>(score2, Wu, user_prop, B);

  launch_zero(out + (size_t)B * 128, 1, stream);   // L2 accumulator slot
  final_score<<<B, 128, 0, stream>>>(user, item, user_emb, item_emb,
                                     user_prop, item_prop, score2, out, B);
}
